// StageNet_14585708937719
// MI455X (gfx1250) — matrix operand order repaired and checked by reading, not compiled, not run on hardware
//
#include <hip/hip_runtime.h>
#include <hip/hip_bf16.h>

// ---------------------------------------------------------------------------
// Fused MVS plane-sweep + 3D-conv(1 out-ch) + softmax depth/conf for gfx1250.
// The 126MB cost volume is never materialized: each block owns a 16x4 spatial
// tile, loops depth slices, builds halo'd 32-channel volume vectors in regs,
// and channel-reduces them to the 27 conv taps with v_wmma_f32_16x16x32_f16
// (K=32 == C in one instruction). Tap sums per kd accumulate into an LDS cost
// row per (h,w); softmax/depth/conf finalize in-block.
// Tap rows padded to 36 floats: 16B-aligned rows -> unconditional b128 LDS
// stores for both WMMA result fragments (no divergent guarded stores), and
// gcd(36,64)=4 keeps bank spread for the phase-2 tap reads.
// ---------------------------------------------------------------------------

typedef __attribute__((ext_vector_type(16))) _Float16 v16h;
typedef __attribute__((ext_vector_type(8)))  float    v8f;

#define C_    32
#define D_    48
#define H_    128
#define W_    160
#define HW_   (H_ * W_)        // 20480
#define CHW_  (C_ * H_ * W_)   // 655360
#define TW_   16
#define TH_   4
#define HALO_W 18              // TW_+2
#define HALO_H 6               // TH_+2
#define NPOS  (HALO_W * HALO_H) // 108 halo positions per depth slice
#define NGRP  7                 // ceil(108/16)
#define TPAD  36                // 27 taps, padded so taps 0..31 fit in-row

// --------------------------- kernel 0: projections -------------------------
// proj_matrices: (1, 5, 2, 4, 4).  Compute per-view rot(3x3)+trans(3) of
// src_new @ inv(ref_new) where X_new[:3,:4] = K[:3,:3] @ E[:3,:4], row3=[0001].
__global__ void proj_kernel(const float* __restrict__ proj, float* __restrict__ ws) {
  int v = threadIdx.x;
  if (v >= 4) return;
  auto M = [&](int view, int which, int i, int j) {
    return proj[view * 32 + which * 16 + i * 4 + j];
  };
  float Ar[9], br[3], As[9], bs[3];
  for (int i = 0; i < 3; ++i) {
    for (int j = 0; j < 3; ++j) {
      float sr = 0.f, ss = 0.f;
      for (int k = 0; k < 3; ++k) {
        sr += M(0, 1, i, k) * M(0, 0, k, j);
        ss += M(v + 1, 1, i, k) * M(v + 1, 0, k, j);
      }
      Ar[i * 3 + j] = sr;
      As[i * 3 + j] = ss;
    }
    float sr = 0.f, ss = 0.f;
    for (int k = 0; k < 3; ++k) {
      sr += M(0, 1, i, k) * M(0, 0, k, 3);
      ss += M(v + 1, 1, i, k) * M(v + 1, 0, k, 3);
    }
    br[i] = sr;
    bs[i] = ss;
  }
  // 3x3 inverse of Ar via adjugate
  float c00 = Ar[4] * Ar[8] - Ar[5] * Ar[7];
  float c01 = Ar[5] * Ar[6] - Ar[3] * Ar[8];
  float c02 = Ar[3] * Ar[7] - Ar[4] * Ar[6];
  float det = Ar[0] * c00 + Ar[1] * c01 + Ar[2] * c02;
  float id  = 1.0f / det;
  float Ai[9];
  Ai[0] = c00 * id;
  Ai[1] = (Ar[2] * Ar[7] - Ar[1] * Ar[8]) * id;
  Ai[2] = (Ar[1] * Ar[5] - Ar[2] * Ar[4]) * id;
  Ai[3] = c01 * id;
  Ai[4] = (Ar[0] * Ar[8] - Ar[2] * Ar[6]) * id;
  Ai[5] = (Ar[2] * Ar[3] - Ar[0] * Ar[5]) * id;
  Ai[6] = c02 * id;
  Ai[7] = (Ar[1] * Ar[6] - Ar[0] * Ar[7]) * id;
  Ai[8] = (Ar[0] * Ar[4] - Ar[1] * Ar[3]) * id;
  float bi[3];
  for (int i = 0; i < 3; ++i)
    bi[i] = -(Ai[i * 3 + 0] * br[0] + Ai[i * 3 + 1] * br[1] + Ai[i * 3 + 2] * br[2]);
  // rot = As @ Ai, trans = As @ bi + bs
  for (int i = 0; i < 3; ++i) {
    for (int j = 0; j < 3; ++j) {
      float s = 0.f;
      for (int k = 0; k < 3; ++k) s += As[i * 3 + k] * Ai[k * 3 + j];
      ws[v * 12 + i * 3 + j] = s;
    }
    float t = bs[i];
    for (int k = 0; k < 3; ++k) t += As[i * 3 + k] * bi[k];
    ws[v * 12 + 9 + i] = t;
  }
}

// --------------------------- kernel 1: fused pipeline ----------------------
__global__ __launch_bounds__(128) void fused_kernel(
    const float* __restrict__ ref_feats,  // (4,1,32,128,160)
    const float* __restrict__ src_feats,  // (4,1,32,128,160)
    const float* __restrict__ dvals_g,    // (1,48)
    const float* __restrict__ w_reg,      // (1,32,3,3,3)
    const float* __restrict__ rt,         // ws: 4 views * (rot9 + trans3)
    float* __restrict__ out)              // depth[HW] then conf[HW]
{
  __shared__ float Tl[112 * TPAD];    // tap-reduced volume, halo positions
  __shared__ float costl[64 * 49];    // per-output cost row over depth
  __shared__ float mats[48];
  __shared__ float dv[D_];

  const int tid  = threadIdx.x;
  const int lane = tid & 31;
  const int wave = tid >> 5;
  const int w0 = blockIdx.x * TW_;
  const int h0 = blockIdx.y * TH_;
  const bool hiHalf = (lane >= 16);

  for (int i = tid; i < 64 * 49; i += 128) costl[i] = 0.f;
  if (tid < 48) { mats[tid] = rt[tid]; dv[tid] = dvals_g[tid]; }
  __syncthreads();

  // A fragments: taps(M=27 pad 32) x channels(K=32), weights pre-scaled by
  // 1/4 (view mean).  ISA 16-bit A 16x32 layout: lane<16 holds K {0..7,16..23}
  // interleaved per VGPR; lane>=16 holds K {8..15,24..31}.  M = lane%16.
  v16h A0, A1;
  {
    const int m0 = (lane & 15);
    const int m1 = 16 + (lane & 15);
#pragma unroll
    for (int hi = 0; hi < 16; ++hi) {
      int vg = hi >> 1, hf = hi & 1;
      int c = ((vg < 4) ? 0 : 16) + (hiHalf ? 8 : 0) + 2 * (vg & 3) + hf;
      A0[hi] = (_Float16)(0.25f * w_reg[c * 27 + m0]);
      A1[hi] = (_Float16)((m1 < 27) ? 0.25f * w_reg[c * 27 + m1] : 0.f);
    }
  }

  const int n = lane & 15;           // B column (position within group)
  const int cbase = hiHalf ? 16 : 0; // B rows: lane<16 K={0..7,16..23}, lane>=16 K={8..15,24..31}

  for (int d = 0; d < D_; ++d) {
    const float depth = dv[d];

    // ---- phase 1: per halo position, build 32-ch volume vector, WMMA-reduce
    for (int g = wave; g < NGRP; g += 4) {
      const int pos = g * 16 + n;
      const int hh = pos / HALO_W;
      const int ww = pos % HALO_W;
      const int h = h0 - 1 + hh;
      const int w = w0 - 1 + ww;
      const bool valid = (pos < NPOS) && (h >= 0) && (h < H_) && (w >= 0) && (w < W_);

      float vol[16];
#pragma unroll
      for (int i = 0; i < 16; ++i) vol[i] = 0.f;

      if (valid) {
        const float fx = (float)w, fy = (float)h;
        for (int v = 0; v < 4; ++v) {
          const float* Mv = &mats[v * 12];
          float pz = (Mv[6] * fx + Mv[7] * fy + Mv[8]) * depth + Mv[11];
          float px = ((Mv[0] * fx + Mv[1] * fy + Mv[2]) * depth + Mv[9]) / pz;
          float py = ((Mv[3] * fx + Mv[4] * fy + Mv[5]) * depth + Mv[10]) / pz;
          float x0 = floorf(px), y0 = floorf(py);
          float wx = px - x0, wy = py - y0;
          int xi = (int)x0, yi = (int)y0;

          float wt[4];
          int off[4];
#pragma unroll
          for (int cy = 0; cy < 2; ++cy) {
#pragma unroll
            for (int cx = 0; cx < 2; ++cx) {
              int xc = xi + cx, yc = yi + cy;
              bool ok = (xc >= 0) && (xc < W_) && (yc >= 0) && (yc < H_);
              int xcc = xc < 0 ? 0 : (xc > W_ - 1 ? W_ - 1 : xc);
              int ycc = yc < 0 ? 0 : (yc > H_ - 1 ? H_ - 1 : yc);
              off[cy * 2 + cx] = ycc * W_ + xcc;
              float wxx = cx ? wx : (1.f - wx);
              float wyy = cy ? wy : (1.f - wy);
              wt[cy * 2 + cx] = ok ? wxx * wyy : 0.f;
            }
          }
          const float* sf = src_feats + v * CHW_;
          const float* rf = ref_feats + v * CHW_ + h * W_ + w;
#pragma unroll
          for (int i = 0; i < 16; ++i) {
            const int c = (i < 8 ? 0 : 8) + (hiHalf ? 8 : 0) + i;
            const float* sc = sf + c * HW_;
            float warp = wt[0] * sc[off[0]] + wt[1] * sc[off[1]] +
                         wt[2] * sc[off[2]] + wt[3] * sc[off[3]];
            vol[i] = fmaf(rf[c * HW_], warp, vol[i]);
          }
        }
      }

      // B fragment (32x16 f16): lanes 0-15 K={0..7,16..23}, lanes 16-31 K={8..15,24..31}
      v16h B;
#pragma unroll
      for (int i = 0; i < 16; ++i) B[i] = (_Float16)vol[i];

      v8f zero = {0.f, 0.f, 0.f, 0.f, 0.f, 0.f, 0.f, 0.f};
      // EXEC is all-ones here (per-lane invalidity handled via zeroed data)
      v8f t0 = __builtin_amdgcn_wmma_f32_16x16x32_f16(
          false, A0, false, B, (short)0, zero, false, false);
      v8f t1 = __builtin_amdgcn_wmma_f32_16x16x32_f16(
          false, A1, false, B, (short)0, zero, false, false);

      // D layout: lane<16 -> rows M=r, lane>=16 -> rows M=8+r; N = lane%16.
      // Row stride 36 floats (144B = 9*16B): all 16 stores in-row and
      // 16B-aligned -> straight b128 stores, no guarded/divergent path.
      float* trow = &Tl[pos * TPAD];
      const int mb = hiHalf ? 8 : 0;
#pragma unroll
      for (int r = 0; r < 8; ++r) trow[mb + r] = t0[r];
#pragma unroll
      for (int r = 0; r < 8; ++r) trow[16 + mb + r] = t1[r];
    }
    __syncthreads();

    // ---- phase 2: 9 spatial taps per kd -> cost[d-1], cost[d], cost[d+1]
    if (tid < 64) {
      const int oh = tid >> 4, ow = tid & 15;
      float p0 = 0.f, p1 = 0.f, p2 = 0.f;
#pragma unroll
      for (int kh = 0; kh < 3; ++kh) {
#pragma unroll
        for (int kw = 0; kw < 3; ++kw) {
          const float* trow = &Tl[((oh + kh) * HALO_W + (ow + kw)) * TPAD];
          const int t = kh * 3 + kw;
          p0 += trow[t];        // kd = 0 taps -> contributes to cost[d+1]
          p1 += trow[9 + t];    // kd = 1 taps -> cost[d]
          p2 += trow[18 + t];   // kd = 2 taps -> cost[d-1]
        }
      }
      float* crow = &costl[tid * 49];
      crow[d] += p1;
      if (d + 1 < D_) crow[d + 1] += p0;
      if (d - 1 >= 0) crow[d - 1] += p2;
    }
    __syncthreads();
  }

  // ---- finalize: softmax over D, expected depth, conf window
  if (tid < 64) {
    const int oh = tid >> 4, ow = tid & 15;
    float* crow = &costl[tid * 49];
    float mx = -1e30f;
    for (int d = 0; d < D_; ++d) mx = fmaxf(mx, crow[d]);
    float s = 0.f;
    for (int d = 0; d < D_; ++d) {
      float e = __expf(crow[d] - mx);
      crow[d] = e;
      s += e;
    }
    const float inv = 1.f / s;
    float depth = 0.f, didx = 0.f;
    for (int d = 0; d < D_; ++d) {
      float p = crow[d] * inv;
      crow[d] = p;
      depth += p * dv[d];
      didx += p * (float)d;
    }
    int idx = (int)didx;
    idx = idx < 0 ? 0 : (idx > D_ - 1 ? D_ - 1 : idx);
    float conf = 0.f;
    for (int j = idx - 1; j <= idx + 2; ++j)
      if (j >= 0 && j < D_) conf += crow[j];
    const int h = h0 + oh, w = w0 + ow;
    out[h * W_ + w] = depth;
    out[HW_ + h * W_ + w] = conf;
  }
}

// ---------------------------------------------------------------------------
extern "C" void kernel_launch(void* const* d_in, const int* in_sizes, int n_in,
                              void* d_out, int out_size, void* d_ws, size_t ws_size,
                              hipStream_t stream) {
  (void)in_sizes; (void)n_in; (void)out_size; (void)ws_size;
  const float* ref_feats = (const float*)d_in[0];
  const float* src_feats = (const float*)d_in[1];
  const float* proj      = (const float*)d_in[2];
  const float* dvals     = (const float*)d_in[3];
  // d_in[4] = num_depth (48, compile-time constant here)
  const float* w_reg     = (const float*)d_in[5];
  float* ws  = (float*)d_ws;
  float* out = (float*)d_out;

  proj_kernel<<<1, 32, 0, stream>>>(proj, ws);
  dim3 grid(W_ / TW_, H_ / TH_);   // 10 x 32 tiles
  fused_kernel<<<grid, 128, 0, stream>>>(ref_feats, src_feats, dvals, w_reg, ws, out);
}
